// MemTransformerLM_77257871720522
// MI455X (gfx1250) — matrix operand order repaired and checked by reading, not compiled, not run on hardware
//
#include <hip/hip_runtime.h>

// ---------------------------------------------------------------------------
// Transformer-XL forward on MI455X (gfx1250), bf16 WMMA everywhere.
// ~1.1 TFLOP of GEMM work -> compute-bound on the matrix pipe; fused attention
// + fused vocab softmax avoid 0.75 GB of score/logit materialization; layer
// activations (== new_mems) live directly in d_out.
// Round 3: native bf16 converts (hardware cvt, not 6-op emulation); q/kv/r_k
// projections store bf16 so attention fragments are raw dword loads; emb_W
// pre-converted to a bf16 L2-resident copy for the fused vocab softmax.
// ---------------------------------------------------------------------------

#define QLEN_   512
#define MEMLEN_ 512
#define KLEN_   1024
#define BSZ_    8
#define NHEAD_  16
#define DHEAD_  64
#define DMODEL_ 1024
#define HD_     1024
#define DINNER_ 4096
#define NLAYER_ 6
#define NTOKEN_ 32000
#define SCALE_  0.125f          // 1/sqrt(64)

typedef __attribute__((ext_vector_type(16))) __bf16 v16bf;
typedef __attribute__((ext_vector_type(8)))  float  v8f;

union FragU { v16bf v; unsigned int u[8]; };

// Native conversions: clang lowers these to the hardware bf16 cvt ops.
__device__ __forceinline__ unsigned short f2bf(float x) {
  __bf16 h = (__bf16)x;
  return __builtin_bit_cast(unsigned short, h);
}
__device__ __forceinline__ unsigned int pack2(float lo, float hi) {
  return (unsigned int)f2bf(lo) | ((unsigned int)f2bf(hi) << 16);
}
__device__ __forceinline__ float bf2f(unsigned short u) {
  return __uint_as_float((unsigned int)u << 16);
}
__device__ __forceinline__ v8f v8zero() {
  v8f z = {0.f,0.f,0.f,0.f,0.f,0.f,0.f,0.f};
  return z;
}
__device__ __forceinline__ v8f wmma_bf16(v16bf a, v16bf b, v8f c) {
  return __builtin_amdgcn_wmma_f32_16x16x32_bf16(false, a, false, b,
                                                 (short)0, c, false, false);
}

// A fragment (16x32 bf16) from LDS, row-major [m][k], stride_s shorts (even).
// Layout: VGPR v holds K = hi*16 + v*2 (+1) for row m (contiguous, matches B).
__device__ __forceinline__ v16bf frag_a_bf16_lds(const unsigned short* base,
                                                 unsigned stride_s,
                                                 unsigned mlane, unsigned hi,
                                                 unsigned koff) {
  FragU f;
  const unsigned int* row = (const unsigned int*)(base + mlane * stride_s);
#pragma unroll
  for (int v = 0; v < 8; ++v) {
    unsigned klo = koff + hi * 16 + v * 2;
    f.u[v] = row[klo >> 1];
  }
  return f.v;
}

// A fragment from f32 LDS (attention probabilities; 1 cvt-pack per pair).
__device__ __forceinline__ v16bf frag_a_f32_lds(const float* base,
                                                unsigned stride_f,
                                                unsigned mlane, unsigned hi,
                                                unsigned koff) {
  FragU f;
  const float* row = base + mlane * stride_f;
#pragma unroll
  for (int v = 0; v < 8; ++v) {
    unsigned klo = koff + hi * 16 + v * 2;
    f.u[v] = pack2(row[klo], row[klo + 1]);
  }
  return f.v;
}

// B fragment (32x16 bf16): lanes 0-15 hold K 0-15, lanes 16-31 hold K 16-31.
// Global bf16 source with contiguous K: raw dword loads, zero conversion.
__device__ __forceinline__ v16bf frag_b_bf16_g(const unsigned short* pcol,
                                               unsigned kstart) {
  FragU f;
#pragma unroll
  for (int v = 0; v < 8; ++v)
    f.u[v] = *(const unsigned int*)(pcol + kstart + 2 * v);
  return f.v;
}

// B fragment from strided bf16 global source (V in prob@V): u16 pair merge.
__device__ __forceinline__ v16bf frag_b_bf16_strided(const unsigned short* pcol,
                                                     unsigned ldk,
                                                     unsigned kstart) {
  FragU f;
#pragma unroll
  for (int v = 0; v < 8; ++v) {
    unsigned lo = pcol[(kstart + 2 * v) * ldk];
    unsigned hi = pcol[(kstart + 2 * v + 1) * ldk];
    f.u[v] = lo | (hi << 16);
  }
  return f.v;
}

// B fragment from LDS stored transposed [n][k].
__device__ __forceinline__ v16bf frag_b_bf16_ldsT(const unsigned short* basecol,
                                                  unsigned kbase) {
  FragU f;
  const unsigned int* row = (const unsigned int*)basecol;
#pragma unroll
  for (int v = 0; v < 8; ++v) f.u[v] = row[(kbase >> 1) + v];
  return f.v;
}

// ---------------------------------------------------------------------------
// Generic bf16 GEMM: C[M,N] = relu?(A[M,K] @ B[K,N] + bias + resid)
// 256 threads (8 waves), 128x128 block tile, BLK_K=64 staged through 32 KB
// LDS (B transposed to [n][k]); 16 WMMAs per barrier pair; global_prefetch of
// next K-tile. BF16OUT stores C as bf16 for attention-bound tensors.
// Requires M%128==0, N%128==0, K%64==0 (true for every call here).
// ---------------------------------------------------------------------------
template <bool BIAS, bool RES, bool RELU, bool BF16OUT>
__global__ __launch_bounds__(256) void gemm_kernel(
    const float* __restrict__ A, const float* __restrict__ B,
    float* __restrict__ C, int M, int N, int K,
    const float* __restrict__ bias, const float* __restrict__ resid) {
  __shared__ __align__(16) unsigned short lds_a[128 * 64];   // [m][k]
  __shared__ __align__(16) unsigned short lds_b[128 * 64];   // [n][k]

  const unsigned tid  = threadIdx.x;
  const unsigned lane = tid & 31, wave = tid >> 5;
  const unsigned wm = wave >> 1, wn = wave & 1;              // 4x2 wave grid
  const unsigned mlane = lane & 15, hi = lane >> 4, hi8 = hi * 8;
  const unsigned m0 = blockIdx.y * 128, n0 = blockIdx.x * 128;
  const unsigned uN = (unsigned)N, uK = (unsigned)K;

  v8f acc[2][4];
#pragma unroll
  for (int i = 0; i < 2; ++i)
#pragma unroll
    for (int j = 0; j < 4; ++j) acc[i][j] = v8zero();

  const unsigned ra = tid >> 4;            // A: row group 0..15
  const unsigned ka = (tid & 15) * 4;      // A: k offset 0..60
  const unsigned kb = tid >> 5;            // B: k group 0..7
  const unsigned nb = (tid & 31) * 4;      // B: n offset 0..124

  for (unsigned k0 = 0; k0 < uK; k0 += 64) {
    // ---- cooperative stage: A 128x64 row-major, B 64x128 -> [n][k] -------
#pragma unroll
    for (int p = 0; p < 8; ++p) {
      unsigned row = ra + p * 16;
      const float4 t = *(const float4*)(A + (m0 + row) * uK + k0 + ka);
      unsigned int* dst = (unsigned int*)(lds_a + row * 64 + ka);
      dst[0] = pack2(t.x, t.y);
      dst[1] = pack2(t.z, t.w);
    }
#pragma unroll
    for (int p = 0; p < 8; ++p) {
      unsigned kk2 = kb + p * 8;
      const float4 t = *(const float4*)(B + (k0 + kk2) * uN + n0 + nb);
      lds_b[(nb + 0) * 64 + kk2] = f2bf(t.x);
      lds_b[(nb + 1) * 64 + kk2] = f2bf(t.y);
      lds_b[(nb + 2) * 64 + kk2] = f2bf(t.z);
      lds_b[(nb + 3) * 64 + kk2] = f2bf(t.w);
    }
    if (k0 + 64 < uK) {   // pull next K-tile toward L2/L0 (global_prefetch)
      __builtin_prefetch(A + (m0 + ra) * uK + k0 + 64 + ka, 0, 3);
      __builtin_prefetch(B + (k0 + 64 + kb) * uN + n0 + nb, 0, 3);
    }
    __syncthreads();

#pragma unroll
    for (unsigned ks = 0; ks < 2; ++ks) {
      v16bf af[2];
#pragma unroll
      for (unsigned mt = 0; mt < 2; ++mt)
        af[mt] = frag_a_bf16_lds(lds_a + (wm * 32 + mt * 16) * 64, 64,
                                 mlane, hi, ks * 32);
#pragma unroll
      for (unsigned nt = 0; nt < 4; ++nt) {
        v16bf bf = frag_b_bf16_ldsT(lds_b + (wn * 64 + nt * 16 + mlane) * 64,
                                    ks * 32 + hi * 16);
        acc[0][nt] = wmma_bf16(af[0], bf, acc[0][nt]);
        acc[1][nt] = wmma_bf16(af[1], bf, acc[1][nt]);
      }
    }
    __syncthreads();
  }

#pragma unroll
  for (unsigned mt = 0; mt < 2; ++mt) {
#pragma unroll
    for (unsigned nt = 0; nt < 4; ++nt) {
      unsigned col = n0 + wn * 64 + nt * 16 + mlane;
      float bv = BIAS ? bias[col] : 0.f;
#pragma unroll
      for (int v = 0; v < 8; ++v) {
        unsigned row = m0 + wm * 32 + mt * 16 + v + hi8;
        float val = acc[mt][nt][v] + bv;
        if (RES) val += resid[row * uN + col];
        if (RELU) val = fmaxf(val, 0.f);
        if (BF16OUT) ((unsigned short*)C)[row * uN + col] = f2bf(val);
        else         C[row * uN + col] = val;
      }
    }
  }
}

// ---------------------------------------------------------------------------
// Fused relative attention, one block per (32-query tile, b, h).
// q/kv/rk arrive as bf16 (produced by BF16OUT GEMMs) -> fragments are raw
// dword loads. AC WMMA -> LDS; BD WMMA with rel-shift scatter-add
// (j = r-511+i, bijective per row so race-free); masked softmax in LDS;
// prob@V WMMA from LDS probs. Dynamic LDS ~137 KB -> 2 workgroups/WGP.
// ---------------------------------------------------------------------------
__global__ __launch_bounds__(256) void attn_kernel(
    const unsigned short* __restrict__ q,    // [qlen, bsz, HD] bf16
    const unsigned short* __restrict__ kv,   // [klen, bsz, 2*HD] bf16 (k | v)
    const unsigned short* __restrict__ rk,   // [klen, HD] bf16
    const float* __restrict__ rwb,           // [HD]
    const float* __restrict__ rrb,           // [HD]
    float* __restrict__ vec) {               // [qlen, bsz, HD] f32
  extern __shared__ char smem[];
  float* sh_score = (float*)smem;                          // 32 x 1024
  float* sh_red   = sh_score + 32 * KLEN_;                 // 32 x 8
  unsigned short* sh_qw = (unsigned short*)(sh_red + 32 * 8);   // 32 x 64
  unsigned short* sh_qr = sh_qw + 32 * 64;

  const unsigned i0 = blockIdx.x * 32;
  const unsigned b  = blockIdx.y;
  const unsigned h  = blockIdx.z;
  const unsigned tid = threadIdx.x;
  const unsigned lane = tid & 31, wave = tid >> 5;
  const unsigned mlane = lane & 15, hi = lane >> 4, hi8 = hi * 8;

  // stage (q + r_w_bias) and (q + r_r_bias) as bf16
  for (unsigned idx = tid; idx < 32 * 64; idx += 256) {
    unsigned i = idx >> 6, d = idx & 63;
    float qv = bf2f(q[((i0 + i) * BSZ_ + b) * HD_ + h * 64 + d]);
    sh_qw[idx] = f2bf(qv + rwb[h * 64 + d]);
    sh_qr[idx] = f2bf(qv + rrb[h * 64 + d]);
  }
  __syncthreads();

  v16bf aw[2][2], ar[2][2];
#pragma unroll
  for (unsigned mt = 0; mt < 2; ++mt)
#pragma unroll
    for (unsigned ks = 0; ks < 2; ++ks) {
      aw[mt][ks] = frag_a_bf16_lds(sh_qw + mt * 16 * 64, 64, mlane, hi, ks * 32);
      ar[mt][ks] = frag_a_bf16_lds(sh_qr + mt * 16 * 64, 64, mlane, hi, ks * 32);
    }

  // ---- AC = (q+rwb) . k  -> sh_score[i][j] -------------------------------
  for (unsigned jt8 = 0; jt8 < 8; ++jt8) {
    unsigned jt = wave * 8 + jt8;                 // 0..63 key tiles
    unsigned j  = jt * 16 + mlane;
    const unsigned short* pk = kv + (j * BSZ_ + b) * (2 * HD_) + h * 64;
    v8f a0 = v8zero(), a1 = v8zero();
#pragma unroll
    for (unsigned ks = 0; ks < 2; ++ks) {
      v16bf bf = frag_b_bf16_g(pk, hi * 16 + ks * 32);
      a0 = wmma_bf16(aw[0][ks], bf, a0);
      a1 = wmma_bf16(aw[1][ks], bf, a1);
    }
#pragma unroll
    for (int v = 0; v < 8; ++v) {
      sh_score[(v + hi8) * KLEN_ + jt * 16 + mlane]      = a0[v];
      sh_score[(16 + v + hi8) * KLEN_ + jt * 16 + mlane] = a1[v];
    }
  }
  __syncthreads();

  // ---- BD = (q+rrb) . r_k with TXL rel-shift: score[i][r-511+i] += bd ----
  for (unsigned rt8 = 0; rt8 < 8; ++rt8) {
    unsigned rt = wave * 8 + rt8;
    int rr = (int)(rt * 16 + mlane);
    const unsigned short* pr = rk + (unsigned)rr * HD_ + h * 64;
    v8f a0 = v8zero(), a1 = v8zero();
#pragma unroll
    for (unsigned ks = 0; ks < 2; ++ks) {
      v16bf bf = frag_b_bf16_g(pr, hi * 16 + ks * 32);
      a0 = wmma_bf16(ar[0][ks], bf, a0);
      a1 = wmma_bf16(ar[1][ks], bf, a1);
    }
#pragma unroll
    for (int v = 0; v < 8; ++v) {
      int il0 = v + (int)hi8;
      int j0  = rr - (QLEN_ - 1) + (int)i0 + il0;
      if (j0 >= 0 && j0 < KLEN_) sh_score[il0 * KLEN_ + j0] += a0[v];
      int il1 = 16 + v + (int)hi8;
      int j1  = rr - (QLEN_ - 1) + (int)i0 + il1;
      if (j1 >= 0 && j1 < KLEN_) sh_score[il1 * KLEN_ + j1] += a1[v];
    }
  }
  __syncthreads();

  // ---- masked softmax over j (8 threads per row) -------------------------
  {
    unsigned row = tid >> 3, sub = tid & 7;
    unsigned jmax = i0 + row + MEMLEN_;      // inclusive causal+memory bound
    float* srow = sh_score + row * KLEN_;
    float mx = -3.0e38f;
    for (unsigned j = sub; j < KLEN_; j += 8)
      if (j <= jmax) mx = fmaxf(mx, srow[j] * SCALE_);
    sh_red[row * 8 + sub] = mx;
    __syncthreads();
    float m = -3.0e38f;
#pragma unroll
    for (int t = 0; t < 8; ++t) m = fmaxf(m, sh_red[row * 8 + t]);
    __syncthreads();
    float s = 0.f;
    for (unsigned j = sub; j < KLEN_; j += 8) {
      float p = 0.f;
      if (j <= jmax) p = __expf(srow[j] * SCALE_ - m);
      srow[j] = p;
      s += p;
    }
    sh_red[row * 8 + sub] = s;
    __syncthreads();
    float ssum = 0.f;
#pragma unroll
    for (int t = 0; t < 8; ++t) ssum += sh_red[row * 8 + t];
    float inv = 1.f / ssum;
    for (unsigned j = sub; j < KLEN_; j += 8) srow[j] *= inv;
  }
  __syncthreads();

  // ---- vec = prob @ V; one 16x16 tile per wave (2 M x 4 N) ---------------
  {
    unsigned mt = wave >> 2, ntv = wave & 3;
    v8f acc = v8zero();
    const unsigned short* vcol =
        kv + b * (2 * HD_) + HD_ + h * 64 + ntv * 16 + mlane;
    const unsigned ldv = BSZ_ * 2 * HD_;
    for (unsigned kk = 0; kk < KLEN_; kk += 32) {
      v16bf af = frag_a_f32_lds(sh_score + (mt * 16) * KLEN_, KLEN_, mlane, hi, kk);
      v16bf bf = frag_b_bf16_strided(vcol, ldv, kk + hi * 16);
      acc = wmma_bf16(af, bf, acc);
    }
#pragma unroll
    for (int v = 0; v < 8; ++v) {
      unsigned il = mt * 16 + v + hi8;
      vec[((i0 + il) * BSZ_ + b) * HD_ + h * 64 + ntv * 16 + mlane] = acc[v];
    }
  }
}

// ---------------------------------------------------------------------------
// Residual-sum already in x; y = LN(x)*g + b, one block per 1024-wide row.
// ---------------------------------------------------------------------------
__global__ __launch_bounds__(256) void ln_kernel(
    const float* __restrict__ x, const float* __restrict__ g,
    const float* __restrict__ bta, float* __restrict__ y) {
  __shared__ float s1[256], s2[256];
  const unsigned row = blockIdx.x, tid = threadIdx.x;
  const float* xr = x + row * (unsigned)DMODEL_;
  float a = 0.f, sq = 0.f, vals[4];
#pragma unroll
  for (int e = 0; e < 4; ++e) {
    float v = xr[tid + 256 * e];
    vals[e] = v; a += v; sq += v * v;
  }
  s1[tid] = a; s2[tid] = sq;
  __syncthreads();
  for (unsigned off = 128; off > 0; off >>= 1) {
    if (tid < off) { s1[tid] += s1[tid + off]; s2[tid] += s2[tid + off]; }
    __syncthreads();
  }
  float mean = s1[0] * (1.f / DMODEL_);
  float var  = s2[0] * (1.f / DMODEL_) - mean * mean;
  float rstd = rsqrtf(var + 1e-5f);
#pragma unroll
  for (int e = 0; e < 4; ++e) {
    unsigned d = tid + 256 * e;
    y[row * (unsigned)DMODEL_ + d] = (vals[e] - mean) * rstd * g[d] + bta[d];
  }
}

// Sinusoidal relative positions: pos[r][d], pos value = klen-1-r.
__global__ void pos_kernel(float* __restrict__ pos) {
  unsigned idx = blockIdx.x * blockDim.x + threadIdx.x;
  unsigned r = idx >> 10, d = idx & 1023;
  float p = (float)(KLEN_ - 1 - (int)r);
  unsigned j = (d < 512) ? d : d - 512;
  float invf = __expf(-(float)j * (9.210340371976184f / 512.0f));  // ln(1e4)/512
  float ang = p * invf;
  pos[idx] = (d < 512) ? sinf(ang) : cosf(ang);
}

// word_emb gather -> new_mems[0] slot (doubles as layer-0 input).
__global__ void embed_kernel(const int* __restrict__ data,
                             const float* __restrict__ embW,
                             float* __restrict__ out) {
  const unsigned row = blockIdx.x, tid = threadIdx.x;
  const float* src = embW + (unsigned)data[row] * DMODEL_;
  float* dst = out + row * (unsigned)DMODEL_;
#pragma unroll
  for (int e = 0; e < 4; ++e) dst[tid + 256 * e] = src[tid + 256 * e];
}

// One-shot f32 -> bf16 tensor copy (emb_W): one float4 -> 2 dwords per thread.
__global__ void cvt_bf16_kernel(const float* __restrict__ in,
                                unsigned short* __restrict__ out) {
  unsigned idx = blockIdx.x * blockDim.x + threadIdx.x;
  const float4 t = *(const float4*)(in + (size_t)idx * 4);
  unsigned int* dst = (unsigned int*)(out + (size_t)idx * 4);
  dst[0] = pack2(t.x, t.y);
  dst[1] = pack2(t.z, t.w);
}

// ---------------------------------------------------------------------------
// Fused tied-softmax loss: per 32-row block, stream 500 vocab tiles of 64
// through WMMA with an online logsumexp. emb_W arrives as a 64 MB bf16 copy
// -> fully L2-resident (192 MB L2) and fragments are raw dword loads.
// ---------------------------------------------------------------------------
__global__ __launch_bounds__(256) void loss_kernel(
    const float* __restrict__ core,            // [4096, 1024] f32
    const unsigned short* __restrict__ embB,   // [32000, 1024] bf16
    const int* __restrict__ target,            // [4096]
    float* __restrict__ loss_out) {
  extern __shared__ char smem[];
  float* sh_tile = (float*)smem;                              // 32 x 64
  unsigned short* sh_core = (unsigned short*)(sh_tile + 32 * 64); // 32 x 1024

  const unsigned m0 = blockIdx.x * 32;
  const unsigned tid = threadIdx.x;
  const unsigned lane = tid & 31, wave = tid >> 5;
  const unsigned mlane = lane & 15, hi = lane >> 4, hi8 = hi * 8;
  const unsigned mt = wave & 1, nt4 = wave >> 1;

  for (unsigned idx = tid; idx < 32 * 1024; idx += 256) {
    unsigned r = idx >> 10, d = idx & 1023;
    sh_core[idx] = f2bf(core[(m0 + r) * (unsigned)DMODEL_ + d]);
  }
  __syncthreads();

  float rm = -3.0e38f, rs = 0.f, rtl = 0.f;
  int tg = (tid < 32) ? target[m0 + tid] : 0;

  for (int v0 = 0; v0 < NTOKEN_; v0 += 64) {
    v8f acc = v8zero();
    unsigned col = (unsigned)v0 + nt4 * 16 + mlane;
    const unsigned short* prow = embB + col * (unsigned)DMODEL_;
    for (unsigned kk = 0; kk < DMODEL_; kk += 32) {
      v16bf af = frag_a_bf16_lds(sh_core + (mt * 16) * 1024, 1024, mlane, hi, kk);
      v16bf bf = frag_b_bf16_g(prow, kk + hi * 16);
      acc = wmma_bf16(af, bf, acc);
    }
#pragma unroll
    for (int v = 0; v < 8; ++v)
      sh_tile[(mt * 16 + v + hi8) * 64 + nt4 * 16 + mlane] = acc[v];
    __syncthreads();

    if (tid < 32) {
      const float* trow = sh_tile + tid * 64;
      float tmax = -3.0e38f;
#pragma unroll 8
      for (int n = 0; n < 64; ++n) tmax = fmaxf(tmax, trow[n]);
      float mnew = fmaxf(rm, tmax);
      float s = rs * __expf(rm - mnew);
#pragma unroll 8
      for (int n = 0; n < 64; ++n) s += __expf(trow[n] - mnew);
      rm = mnew; rs = s;
      if (tg >= v0 && tg < v0 + 64) rtl = trow[tg - v0];
    }
    __syncthreads();
  }
  if (tid < 32) loss_out[m0 + tid] = -(rtl - (rm + logf(rs)));
}

// ---------------------------------------------------------------------------
static inline void launch_gemm(const float* A, const float* B, void* C,
                               int M, int N, int K, const float* bias,
                               const float* resid, bool relu, bool bf16out,
                               hipStream_t stream) {
  dim3 g((unsigned)(N / 128), (unsigned)(M / 128)), b(256);
  float* Cf = (float*)C;
  if (bf16out)
    gemm_kernel<false, false, false, true ><<<g, b, 0, stream>>>(A, B, Cf, M, N, K, bias, resid);
  else if (bias && relu)
    gemm_kernel<true,  false, true,  false><<<g, b, 0, stream>>>(A, B, Cf, M, N, K, bias, resid);
  else if (bias && resid)
    gemm_kernel<true,  true,  false, false><<<g, b, 0, stream>>>(A, B, Cf, M, N, K, bias, resid);
  else if (resid)
    gemm_kernel<false, true,  false, false><<<g, b, 0, stream>>>(A, B, Cf, M, N, K, bias, resid);
  else
    gemm_kernel<false, false, false, false><<<g, b, 0, stream>>>(A, B, Cf, M, N, K, bias, resid);
}

extern "C" void kernel_launch(void* const* d_in, const int* in_sizes, int n_in,
                              void* d_out, int out_size, void* d_ws,
                              size_t ws_size, hipStream_t stream) {
  (void)in_sizes; (void)n_in; (void)out_size; (void)ws_size;
  const int*   data   = (const int*)d_in[0];
  const int*   target = (const int*)d_in[1];
  const float* memory = (const float*)d_in[2];
  const float* embW   = (const float*)d_in[3];
  const float* rwb    = (const float*)d_in[4];
  const float* rrb    = (const float*)d_in[5];
  const float* Wq     = (const float*)d_in[6];
  const float* Wkv    = (const float*)d_in[7];
  const float* Wr     = (const float*)d_in[8];
  const float* Wo     = (const float*)d_in[9];
  const float* W1     = (const float*)d_in[10];
  const float* b1     = (const float*)d_in[11];
  const float* W2     = (const float*)d_in[12];
  const float* b2     = (const float*)d_in[13];
  const float* ln1g   = (const float*)d_in[14];
  const float* ln1b   = (const float*)d_in[15];
  const float* ln2g   = (const float*)d_in[16];
  const float* ln2b   = (const float*)d_in[17];

  float* out_loss = (float*)d_out;
  float* out_mems = out_loss + (size_t)QLEN_ * BSZ_;     // [6,512,8,1024]
  const size_t AROWS = (size_t)QLEN_ * BSZ_;             // 4096
  const size_t XROWS = (size_t)KLEN_ * BSZ_;             // 8192
  const size_t CHUNK = (size_t)MEMLEN_ * BSZ_ * DMODEL_; // per-layer slot

  // byte-carved workspace (mixed f32 / bf16), 256 B aligned per buffer
  char* wsb = (char*)d_ws;
  auto carve = [&](size_t bytes) -> char* {
    char* p = wsb;
    wsb += (bytes + 255) & ~(size_t)255;
    return p;
  };
  float*          x_mem = (float*)         carve(XROWS * DMODEL_ * 4);
  unsigned short* qb    = (unsigned short*)carve(AROWS * HD_ * 2);
  unsigned short* kvb   = (unsigned short*)carve(XROWS * 2 * HD_ * 2);
  unsigned short* rkb   = (unsigned short*)carve((size_t)KLEN_ * HD_ * 2);
  float*          posb  = (float*)         carve((size_t)KLEN_ * DMODEL_ * 4);
  float*          vecb  = (float*)         carve(AROWS * HD_ * 4);
  float*          t1    = (float*)         carve(AROWS * DMODEL_ * 4);
  float*          cur   = (float*)         carve(AROWS * DMODEL_ * 4);
  float*          t2    = (float*)         carve(AROWS * DINNER_ * 4);
  float*          coreF = (float*)         carve(AROWS * DMODEL_ * 4);
  unsigned short* embB  = (unsigned short*)carve((size_t)NTOKEN_ * DMODEL_ * 2);

  const dim3 blk(256);
  pos_kernel<<<dim3((KLEN_ * DMODEL_) / 256), blk, 0, stream>>>(posb);
  embed_kernel<<<dim3((unsigned)AROWS), blk, 0, stream>>>(data, embW, out_mems);
  cvt_bf16_kernel<<<dim3((unsigned)((size_t)NTOKEN_ * DMODEL_ / 4 / 256)), blk,
                    0, stream>>>(embW, embB);

  const size_t attn_smem = (size_t)32 * KLEN_ * 4 + 32 * 8 * 4
                         + 2 * (size_t)32 * 64 * 2;      // ~137 KB (< 320 KB/WGP)
  const size_t loss_smem = (size_t)32 * 64 * 4 + (size_t)32 * 1024 * 2;

  for (int L = 0; L < NLAYER_; ++L) {
    const float* core = out_mems + (size_t)L * CHUNK;    // hids[L] == new_mems[L]
    hipMemcpyAsync(x_mem, memory + (size_t)L * CHUNK, CHUNK * sizeof(float),
                   hipMemcpyDeviceToDevice, stream);
    hipMemcpyAsync(x_mem + CHUNK, core, CHUNK * sizeof(float),
                   hipMemcpyDeviceToDevice, stream);

    launch_gemm(core, Wq + (size_t)L * DMODEL_ * HD_, qb,
                (int)AROWS, HD_, DMODEL_, nullptr, nullptr, false, true, stream);
    launch_gemm(x_mem, Wkv + (size_t)L * DMODEL_ * 2 * HD_, kvb,
                (int)XROWS, 2 * HD_, DMODEL_, nullptr, nullptr, false, true, stream);
    launch_gemm(posb, Wr + (size_t)L * DMODEL_ * HD_, rkb,
                KLEN_, HD_, DMODEL_, nullptr, nullptr, false, true, stream);

    attn_kernel<<<dim3(QLEN_ / 32, BSZ_, NHEAD_), blk, attn_smem, stream>>>(
        qb, kvb, rkb, rwb, rrb, vecb);

    launch_gemm(vecb, Wo + (size_t)L * HD_ * DMODEL_, t1,
                (int)AROWS, DMODEL_, HD_, nullptr, core, false, false, stream);
    ln_kernel<<<dim3((unsigned)AROWS), blk, 0, stream>>>(
        t1, ln1g + (size_t)L * DMODEL_, ln1b + (size_t)L * DMODEL_, cur);

    launch_gemm(cur, W1 + (size_t)L * DMODEL_ * DINNER_, t2,
                (int)AROWS, DINNER_, DMODEL_, b1 + (size_t)L * DINNER_,
                nullptr, true, false, stream);
    launch_gemm(t2, W2 + (size_t)L * DINNER_ * DMODEL_, t1,
                (int)AROWS, DMODEL_, DINNER_, b2 + (size_t)L * DMODEL_,
                cur, false, false, stream);

    float* nxt = (L < NLAYER_ - 1) ? (out_mems + (size_t)(L + 1) * CHUNK) : coreF;
    ln_kernel<<<dim3((unsigned)AROWS), blk, 0, stream>>>(
        t1, ln2g + (size_t)L * DMODEL_, ln2b + (size_t)L * DMODEL_, nxt);
  }

  loss_kernel<<<dim3((unsigned)(AROWS / 32)), blk, loss_smem, stream>>>(
      coreF, embB, target, out_loss);
}
